// TemporalAttention_27986006900895
// MI455X (gfx1250) — matrix operand order repaired and checked by reading, not compiled, not run on hardware
//
#include <hip/hip_runtime.h>

// ---------------------------------------------------------------------------
// Problem constants (match reference)
// ---------------------------------------------------------------------------
constexpr int kB  = 4;
constexpr int kS  = 2048;
constexpr int kDI = 512;   // D_IN
constexpr int kD  = 512;   // D_H
constexpr int kH  = 8;
constexpr int kHD = 64;    // head dim
constexpr float kEPS = 1e-5f;

constexpr int kM = kB * kS;          // 8192 rows for projection GEMMs
constexpr float kNEG = -3.0e38f;

// ---------------------------------------------------------------------------
// Vector types
// ---------------------------------------------------------------------------
typedef __attribute__((ext_vector_type(16))) _Float16 v16h;
typedef __attribute__((ext_vector_type(8)))  _Float16 v8h;
typedef __attribute__((ext_vector_type(8)))  float    v8f;
typedef __attribute__((ext_vector_type(4)))  float    v4f;

#define WMMA_F32_F16(a, b, c) \
    __builtin_amdgcn_wmma_f32_16x16x32_f16(false, (a), false, (b), (short)0, (c), false, false)

#define CAT16(lo, hi) \
    __builtin_shufflevector((lo), (hi), 0, 1, 2, 3, 4, 5, 6, 7, 8, 9, 10, 11, 12, 13, 14, 15)

// ---------------------------------------------------------------------------
// Fragment loaders (wave32 layouts per CDNA5 ISA 7.12.2), fully vectorized.
// ---------------------------------------------------------------------------
// A-matrix 16x32 f16, row-major source (leading dim lda, both multiples of 8).
// lane 0-15: M=lane, K={kb..kb+7, 16+kb..16+kb+7} with kb=0; lanes 16-31: kb=8.
// Two contiguous 16B loads per lane.
__device__ __forceinline__ v16h load_a_frag_f16(const _Float16* __restrict__ A,
                                                int lda, int row0, int k0, int lane) {
    int m  = lane & 15;
    int kb = (lane >> 4) << 3;             // 0 or 8
    const _Float16* p = A + (size_t)(row0 + m) * lda + k0 + kb;
    v8h lo = *(const v8h*)(p);
    v8h hi = *(const v8h*)(p + 16);
    return CAT16(lo, hi);
}

// Same A layout, f32 source (attn probs) -> convert in registers. 4x float4 loads.
__device__ __forceinline__ v16h load_a_frag_f32(const float* __restrict__ A,
                                                int lda, int row0, int k0, int lane) {
    int m  = lane & 15;
    int kb = (lane >> 4) << 3;
    const float* p = A + (size_t)(row0 + m) * lda + k0 + kb;
    v4f f0 = *(const v4f*)(p);
    v4f f1 = *(const v4f*)(p + 4);
    v4f f2 = *(const v4f*)(p + 16);
    v4f f3 = *(const v4f*)(p + 20);
    v16h a;
#pragma unroll
    for (int r = 0; r < 4; ++r) {
        a[r]      = (_Float16)f0[r];
        a[4 + r]  = (_Float16)f1[r];
        a[8 + r]  = (_Float16)f2[r];
        a[12 + r] = (_Float16)f3[r];
    }
    return a;
}

// B-matrix 32x16 fragment from a B^T row-major source: element (k,n) lives at
// Mm[(col0+n)*ld + k]. K is contiguous -> two 16B loads per lane.
// lane 0-15: N=lane, K={kb..kb+7, 16+kb..16+kb+7} with kb=0; lanes 16-31: kb=8.
__device__ __forceinline__ v16h load_bT_frag_f16(const _Float16* __restrict__ Mm,
                                                 int ld, int k0, int col0, int lane) {
    int n  = lane & 15;
    int ko = (lane >> 4) << 3;             // 0 or 8
    const _Float16* p = Mm + (size_t)(col0 + n) * ld + k0 + ko;
    v8h lo = *(const v8h*)(p);
    v8h hi = *(const v8h*)(p + 16);
    return CAT16(lo, hi);
}

// ---------------------------------------------------------------------------
// Kernel 0a: f32 -> f16 conversion (elementwise)
// ---------------------------------------------------------------------------
__global__ __launch_bounds__(256) void cvt_f32_to_f16(const float* __restrict__ src,
                                                      _Float16* __restrict__ dst, int n) {
    int i = blockIdx.x * 256 + threadIdx.x;
    if (i < n) dst[i] = (_Float16)src[i];
}

// Kernel 0b: f32 [K,N] -> f16 transposed [N,K] (for weights; coalesced reads)
__global__ __launch_bounds__(256) void cvt_transpose_f16(const float* __restrict__ src,
                                                         _Float16* __restrict__ dst,
                                                         int K, int N) {
    int i = blockIdx.x * 256 + threadIdx.x;
    if (i < K * N) {
        int k = i / N;
        int n = i - k * N;
        dst[(size_t)n * K + k] = (_Float16)src[i];
    }
}

// ---------------------------------------------------------------------------
// Kernel 1: QKV projection GEMM. out = x@W + bias (+ pe), scaled, f16.
// Weights supplied transposed [N,K]. dstMode: 0 -> head layout [B,H,S,HD]
// (q,k); 1 -> transposed head layout [B,H,HD,S] (v, feeds attn@V B-side).
// ---------------------------------------------------------------------------
__global__ __launch_bounds__(256) void qkv_gemm(const _Float16* __restrict__ xh,
                                                const _Float16* __restrict__ WT,
                                                const float* __restrict__ bias,
                                                const float* __restrict__ pe,   // nullptr for v
                                                float scale, int dstMode,
                                                _Float16* __restrict__ dst) {
    const int tilesN = kD / 16;                       // 32
    int wave = blockIdx.x * 8 + (threadIdx.x >> 5);
    int lane = threadIdx.x & 31;
    int mt = wave / tilesN;
    int nt = wave % tilesN;
    int row0 = mt * 16, col0 = nt * 16;

    v8f c = {};
#pragma unroll 4
    for (int k0 = 0; k0 < kDI; k0 += 32) {
        v16h a = load_a_frag_f16(xh, kDI, row0, k0, lane);
        v16h b = load_bT_frag_f16(WT, kDI, k0, col0, lane);
        c = WMMA_F32_F16(a, b, c);
    }

    int n  = lane & 15;
    int mb = (lane >> 4) << 3;
    int col = col0 + n;
    int h  = col >> 6;
    int hd = col & 63;
    float bv = bias[col];
#pragma unroll
    for (int r = 0; r < 8; ++r) {
        int row   = row0 + mb + r;
        int batch = row >> 11;          // / kS
        int s     = row & (kS - 1);     // % kS
        float v = c[r] + bv;
        if (pe) v += pe[(size_t)s * kD + col];
        v *= scale;
        size_t idx = (dstMode == 0)
            ? ((((size_t)batch * kH + h) * kS + s) * kHD + hd)     // [B,H,S,HD]
            : ((((size_t)batch * kH + h) * kHD + hd) * kS + s);    // [B,H,HD,S]
        dst[idx] = (_Float16)v;
    }
}

// ---------------------------------------------------------------------------
// Kernel 2: scores = q @ k^T per (b,h), masked, f32 into attn region of d_out.
// q carries the 1/sqrt(HD) scale. kh is [S,HD] -> B^T loads are contiguous.
// ---------------------------------------------------------------------------
__global__ __launch_bounds__(256) void scores_gemm(const _Float16* __restrict__ qh,
                                                   const _Float16* __restrict__ kh,
                                                   const int* __restrict__ mask,
                                                   float* __restrict__ attn) {
    const int tilesK = kS / 16;                       // 128
    const int perBH  = (kS / 16) * tilesK;            // 16384
    int wave = blockIdx.x * 8 + (threadIdx.x >> 5);
    int lane = threadIdx.x & 31;
    int bh = wave / perBH;
    int t  = wave % perBH;
    int qt = t / tilesK;
    int kt = t % tilesK;

    const _Float16* Q = qh + (size_t)bh * kS * kHD;
    const _Float16* K = kh + (size_t)bh * kS * kHD;
    int row0 = qt * 16, col0 = kt * 16;

    v8f c = {};
#pragma unroll
    for (int k0 = 0; k0 < kHD; k0 += 32) {
        v16h a = load_a_frag_f16(Q, kHD, row0, k0, lane);
        v16h b = load_bT_frag_f16(K, kHD, k0, col0, lane);
        c = WMMA_F32_F16(a, b, c);
    }

    int batch = bh >> 3;                               // / kH
    const int* mrow = mask + (size_t)batch * kS * kS;
    float* out = attn + (size_t)bh * kS * kS;
    int n  = lane & 15;
    int mb = (lane >> 4) << 3;
#pragma unroll
    for (int r = 0; r < 8; ++r) {
        int qrow = row0 + mb + r;
        int kcol = col0 + n;
        float v = c[r];
        if (mrow[(size_t)qrow * kS + kcol] == 0) v = kNEG;
        out[(size_t)qrow * kS + kcol] = v;
    }
}

// ---------------------------------------------------------------------------
// Kernel 3: row softmax in place over S=2048 (one block per row), float4 I/O
// ---------------------------------------------------------------------------
__device__ __forceinline__ float block_reduce_max(float v, float* sbuf) {
    int tid = threadIdx.x;
    sbuf[tid] = v;
    __syncthreads();
    for (int s = 128; s > 0; s >>= 1) {
        if (tid < s) sbuf[tid] = fmaxf(sbuf[tid], sbuf[tid + s]);
        __syncthreads();
    }
    float r = sbuf[0];
    __syncthreads();
    return r;
}
__device__ __forceinline__ float block_reduce_sum(float v, float* sbuf) {
    int tid = threadIdx.x;
    sbuf[tid] = v;
    __syncthreads();
    for (int s = 128; s > 0; s >>= 1) {
        if (tid < s) sbuf[tid] += sbuf[tid + s];
        __syncthreads();
    }
    float r = sbuf[0];
    __syncthreads();
    return r;
}

__global__ __launch_bounds__(256) void softmax_rows(float* __restrict__ attn) {
    __shared__ float sbuf[256];
    v4f* row = (v4f*)(attn + (size_t)blockIdx.x * kS);
    const int nvec = kS / 4;                           // 512 float4s
    int tid = threadIdx.x;

    float m = kNEG;
    for (int i = tid; i < nvec; i += 256) {
        v4f q = row[i];
        m = fmaxf(m, fmaxf(fmaxf(q.x, q.y), fmaxf(q.z, q.w)));
    }
    m = block_reduce_max(m, sbuf);

    float s = 0.0f;
    for (int i = tid; i < nvec; i += 256) {
        v4f q = row[i];
        q.x = __expf(q.x - m);
        q.y = __expf(q.y - m);
        q.z = __expf(q.z - m);
        q.w = __expf(q.w - m);
        row[i] = q;
        s += q.x + q.y + q.z + q.w;
    }
    s = block_reduce_sum(s, sbuf);
    float inv = 1.0f / s;
    for (int i = tid; i < nvec; i += 256) {
        v4f q = row[i];
        q.x *= inv; q.y *= inv; q.z *= inv; q.w *= inv;
        row[i] = q;
    }
}

// ---------------------------------------------------------------------------
// Kernel 4: O = attn @ V per (b,h). One wave owns a full 16x64 output block
// (4 accumulators) so the 537MB prob matrix is read exactly once.
// vT is [B,H,HD,S] so B-fragments are contiguous along K.
// ---------------------------------------------------------------------------
__global__ __launch_bounds__(256) void av_gemm(const float* __restrict__ attn,
                                               const _Float16* __restrict__ vT,
                                               _Float16* __restrict__ oh) {
    const int tilesM = kS / 16;                       // 128
    int wave = blockIdx.x * 8 + (threadIdx.x >> 5);
    int lane = threadIdx.x & 31;
    int bh = wave / tilesM;
    int mt = wave % tilesM;

    const float*    P = attn + (size_t)bh * kS * kS;
    const _Float16* V = vT   + (size_t)bh * kS * kHD; // [HD=64, S] rows
    int row0 = mt * 16;

    v8f c0 = {}, c1 = {}, c2 = {}, c3 = {};
    for (int k0 = 0; k0 < kS; k0 += 32) {
        v16h a  = load_a_frag_f32(P, kS, row0, k0, lane);
        v16h b0 = load_bT_frag_f16(V, kS, k0, 0,  lane);
        v16h b1 = load_bT_frag_f16(V, kS, k0, 16, lane);
        v16h b2 = load_bT_frag_f16(V, kS, k0, 32, lane);
        v16h b3 = load_bT_frag_f16(V, kS, k0, 48, lane);
        c0 = WMMA_F32_F16(a, b0, c0);
        c1 = WMMA_F32_F16(a, b1, c1);
        c2 = WMMA_F32_F16(a, b2, c2);
        c3 = WMMA_F32_F16(a, b3, c3);
    }

    int batch = bh >> 3;
    int h     = bh & 7;
    int n  = lane & 15;
    int mb = (lane >> 4) << 3;
#pragma unroll
    for (int r = 0; r < 8; ++r) {
        int srow = row0 + mb + r;
        _Float16* orow = oh + ((size_t)batch * kS + srow) * kD + h * kHD + n;
        orow[0]  = (_Float16)c0[r];
        orow[16] = (_Float16)c1[r];
        orow[32] = (_Float16)c2[r];
        orow[48] = (_Float16)c3[r];
    }
}

// ---------------------------------------------------------------------------
// Kernel 5: proj = oh @ Wo + bo  (Wo supplied transposed; f32 to workspace)
// ---------------------------------------------------------------------------
__global__ __launch_bounds__(256) void proj_gemm(const _Float16* __restrict__ oh,
                                                 const _Float16* __restrict__ WoT,
                                                 const float* __restrict__ bo,
                                                 float* __restrict__ proj) {
    const int tilesN = kD / 16;                       // 32
    int wave = blockIdx.x * 8 + (threadIdx.x >> 5);
    int lane = threadIdx.x & 31;
    int mt = wave / tilesN;
    int nt = wave % tilesN;
    int row0 = mt * 16, col0 = nt * 16;

    v8f c = {};
#pragma unroll 4
    for (int k0 = 0; k0 < kD; k0 += 32) {
        v16h a = load_a_frag_f16(oh, kD, row0, k0, lane);
        v16h b = load_bT_frag_f16(WoT, kD, k0, col0, lane);
        c = WMMA_F32_F16(a, b, c);
    }

    int n  = lane & 15;
    int mb = (lane >> 4) << 3;
    int col = col0 + n;
    float bv = bo[col];
#pragma unroll
    for (int r = 0; r < 8; ++r) {
        int row = row0 + mb + r;
        proj[(size_t)row * kD + col] = c[r] + bv;
    }
}

// ---------------------------------------------------------------------------
// Kernel 6: layernorm per row of 512 -> d_out
// ---------------------------------------------------------------------------
__global__ __launch_bounds__(256) void layernorm_rows(const float* __restrict__ proj,
                                                      const float* __restrict__ gamma,
                                                      const float* __restrict__ beta,
                                                      float* __restrict__ out) {
    __shared__ float sbuf[256];
    const float* p = proj + (size_t)blockIdx.x * kD;
    float* o = out + (size_t)blockIdx.x * kD;
    int tid = threadIdx.x;

    float s0 = 0.0f, s1 = 0.0f;
    float a = p[tid], b = p[tid + 256];
    s0 = a + b;
    float mean = block_reduce_sum(s0, sbuf) * (1.0f / kD);

    float da = a - mean, db = b - mean;
    s1 = da * da + db * db;
    float var = block_reduce_sum(s1, sbuf) * (1.0f / kD);
    float rstd = rsqrtf(var + kEPS);

    o[tid]       = da * rstd * gamma[tid]       + beta[tid];
    o[tid + 256] = db * rstd * gamma[tid + 256] + beta[tid + 256];
}

// ---------------------------------------------------------------------------
// Host launch
// ---------------------------------------------------------------------------
extern "C" void kernel_launch(void* const* d_in, const int* in_sizes, int n_in,
                              void* d_out, int out_size, void* d_ws, size_t ws_size,
                              hipStream_t stream) {
    (void)in_sizes; (void)n_in; (void)out_size; (void)ws_size;

    const float* x    = (const float*)d_in[0];
    const int*   mask = (const int*)  d_in[1];
    const float* Wq   = (const float*)d_in[2];
    const float* bq   = (const float*)d_in[3];
    const float* Wk   = (const float*)d_in[4];
    const float* bk   = (const float*)d_in[5];
    const float* Wv   = (const float*)d_in[6];
    const float* bv   = (const float*)d_in[7];
    const float* pe   = (const float*)d_in[8];
    const float* Wo   = (const float*)d_in[9];
    const float* bo   = (const float*)d_in[10];
    const float* gamma= (const float*)d_in[11];
    const float* beta = (const float*)d_in[12];

    float* out_main = (float*)d_out;                              // [B,S,D_H]
    float* attn     = out_main + (size_t)kB * kS * kD;            // [B,H,S,S]

    // workspace carve-up
    char* ws = (char*)d_ws;
    size_t off = 0;
    _Float16* xh   = (_Float16*)(ws + off); off += (size_t)kM * kDI * 2;
    _Float16* WqT  = (_Float16*)(ws + off); off += (size_t)kDI * kD * 2;
    _Float16* WkT  = (_Float16*)(ws + off); off += (size_t)kDI * kD * 2;
    _Float16* WvT  = (_Float16*)(ws + off); off += (size_t)kDI * kD * 2;
    _Float16* WoT  = (_Float16*)(ws + off); off += (size_t)kD  * kD * 2;
    _Float16* qh   = (_Float16*)(ws + off); off += (size_t)kB * kH * kS * kHD * 2;
    _Float16* kh   = (_Float16*)(ws + off); off += (size_t)kB * kH * kS * kHD * 2;
    _Float16* vT   = (_Float16*)(ws + off); off += (size_t)kB * kH * kS * kHD * 2;
    _Float16* oh   = (_Float16*)(ws + off); off += (size_t)kM * kD * 2;
    float*    proj = (float*)   (ws + off); off += (size_t)kM * kD * 4;   // ~61 MB

    // 0) conversions (+ weight transposes)
    {
        int n = kM * kDI;
        cvt_f32_to_f16<<<(n + 255) / 256, 256, 0, stream>>>(x, xh, n);
        n = kDI * kD;
        int blks = (n + 255) / 256;
        cvt_transpose_f16<<<blks, 256, 0, stream>>>(Wq, WqT, kDI, kD);
        cvt_transpose_f16<<<blks, 256, 0, stream>>>(Wk, WkT, kDI, kD);
        cvt_transpose_f16<<<blks, 256, 0, stream>>>(Wv, WvT, kDI, kD);
        cvt_transpose_f16<<<blks, 256, 0, stream>>>(Wo, WoT, kD,  kD);
    }

    // 1) QKV projections: (8192/16)*(512/16) = 16384 waves -> 2048 blocks
    {
        const float qscale = 0.125f;   // 1/sqrt(64) folded into q
        int blocks = (kM / 16) * (kD / 16) / 8;
        qkv_gemm<<<blocks, 256, 0, stream>>>(xh, WqT, bq, pe, qscale, 0, qh);
        qkv_gemm<<<blocks, 256, 0, stream>>>(xh, WkT, bk, pe, 1.0f,  0, kh);
        qkv_gemm<<<blocks, 256, 0, stream>>>(xh, WvT, bv, nullptr, 1.0f, 1, vT);
    }

    // 2) scores: 32 * 128 * 128 = 524288 waves -> 65536 blocks
    {
        int blocks = kB * kH * (kS / 16) * (kS / 16) / 8;
        scores_gemm<<<blocks, 256, 0, stream>>>(qh, kh, mask, attn);
    }

    // 3) softmax: one block per row
    softmax_rows<<<kB * kH * kS, 256, 0, stream>>>(attn);

    // 4) attn @ V: 32 * 128 = 4096 waves -> 512 blocks
    {
        int blocks = kB * kH * (kS / 16) / 8;
        av_gemm<<<blocks, 256, 0, stream>>>(attn, vT, oh);
    }

    // 5) output projection
    {
        int blocks = (kM / 16) * (kD / 16) / 8;
        proj_gemm<<<blocks, 256, 0, stream>>>(oh, WoT, bo, proj);
    }

    // 6) layernorm
    layernorm_rows<<<kM, 256, 0, stream>>>(proj, gamma, beta, out_main);
}
